// ENLSA_21723944583599
// MI455X (gfx1250) — matrix operand order repaired and checked by reading, not compiled, not run on hardware
//
#include <hip/hip_runtime.h>
#include <math.h>

typedef __attribute__((ext_vector_type(16))) _Float16 v16h;
typedef __attribute__((ext_vector_type(8)))  float    v8f;

#define DIMC   512
#define NHEAD  8
#define HD     64
#define NTOK   4096
#define BATCH  2
#define MROWS  (BATCH*NTOK)   // 8192

#if defined(__AMDGCN__) && __has_builtin(__builtin_amdgcn_global_load_async_to_lds_b128) && \
    __has_builtin(__builtin_amdgcn_s_wait_asynccnt)
#define HAVE_ASYNC_LDS 1
#endif

#ifdef HAVE_ASYNC_LDS
typedef int v4i __attribute__((vector_size(4 * sizeof(int))));
typedef __attribute__((address_space(1))) v4i gv4i;   // global (printed as __device__)
typedef __attribute__((address_space(3))) v4i lv4i;   // LDS
static __device__ __forceinline__ void async_cp16(const unsigned short* g, unsigned short* l) {
  __builtin_amdgcn_global_load_async_to_lds_b128((gv4i*)g, (lv4i*)l, 0, 0);
}
static __device__ __forceinline__ void async_wait0() {
  __builtin_amdgcn_s_wait_asynccnt(0);
}
#endif

// cross-lane xor reduce via ds_swizzle (group-of-32 mode, and_mask=0x1f)
#define SWZ_XOR_F(v, m) \
  __builtin_bit_cast(float, __builtin_amdgcn_ds_swizzle(__builtin_bit_cast(int, (v)), ((m) << 10) | 0x1f))

static __device__ __forceinline__ unsigned short f2h(float f) {
  union { _Float16 h; unsigned short u; } c; c.h = (_Float16)f; return c.u;
}

union UF { v16h v; unsigned int u[8]; };

// A-fragment (16xK slice, K-offset koff) from LDS, row-major, ld u16 elements (even).
static __device__ __forceinline__ v16h load_afrag(const unsigned short* lds, int row0,
                                                  int ld, int koff, int lane) {
  UF f;
  const unsigned int* p = (const unsigned int*)lds;
  int base = (row0 + (lane & 15)) * ld + koff + ((lane >> 4) << 3);
#pragma unroll
  for (int j = 0; j < 8; ++j) {
    int kk = ((j & 4) << 2) + ((j & 3) << 1);   // j<4 -> 0..7 ; j>=4 -> 16..23 (lane-half adds 8)
    f.u[j] = p[(base + kk) >> 1];
  }
  return f.v;
}

// B-fragment (Kx16) from LDS laid out [n][k] (k contiguous), ld u16 elements (even).
static __device__ __forceinline__ v16h load_bfrag(const unsigned short* lds, int n0,
                                                  int ld, int koff, int lane) {
  UF f;
  const unsigned int* p = (const unsigned int*)lds;
  int base = (n0 + (lane & 15)) * ld + koff + ((lane >> 4) << 3);
#pragma unroll
  for (int j = 0; j < 8; ++j) {
    int kk = ((j & 4) << 2) + ((j & 3) << 1);
    f.u[j] = p[(base + kk) >> 1];
  }
  return f.v;
}

static __device__ __forceinline__ v8f wmma16(v16h a, v16h b, v8f c) {
  return __builtin_amdgcn_wmma_f32_16x16x32_f16(false, a, false, b, (short)0, c, false, false);
}

static __device__ __forceinline__ v8f vzero8() {
  v8f z;
#pragma unroll
  for (int i = 0; i < 8; ++i) z[i] = 0.0f;
  return z;
}

// ---------------- fp32 -> f16 conversion ----------------
__global__ void cvt_f32_f16(const float* __restrict__ s, unsigned short* __restrict__ d, int n) {
  int i = blockIdx.x * 256 + threadIdx.x;
  if (i < n) d[i] = f2h(s[i]);
}

// ---------------- x [B,C,N] -> xf16 [B,N,C] ----------------
__global__ __launch_bounds__(1024) void xpose_f16(const float* __restrict__ x,
                                                  unsigned short* __restrict__ xf) {
  __shared__ float t[32][33];
  int b = blockIdx.z;
  int n0 = blockIdx.x * 32, c0 = blockIdx.y * 32;
  int tx = threadIdx.x, ty = threadIdx.y;
  t[ty][tx] = x[((size_t)b * DIMC + c0 + ty) * NTOK + n0 + tx];
  __syncthreads();
  xf[((size_t)b * NTOK + n0 + ty) * DIMC + c0 + tx] = f2h(t[tx][ty]);
}

// ---------------- generic f16 WMMA GEMM with fused epilogues ----------------
// MODE 0: QKV scatter -> f16 [B,nh,N,hd], val=(acc+bias)*scale
// MODE 2: O-proj       -> fp32 y[gm*512+gn] = acc+bias
// MODE 3: MLP1 GELU    -> f16  g[gm*N+gn]   = gelu(acc+bias)
// MODE 4: MLP2 + residuals, transposed store -> out[b][gn][n] = acc+bias+y
#define GBM 128
#define GBN 64
#define GLDA 40
#define GLDB 40

template <int MODE>
__global__ __launch_bounds__(256) void gemm16(const unsigned short* __restrict__ A,
                                              const unsigned short* __restrict__ W,
                                              const float* __restrict__ bias,
                                              void* __restrict__ outp,
                                              const float* __restrict__ aux,
                                              int M, int N, int K, float scale) {
  __shared__ __align__(16) unsigned short Alds[GBM * GLDA];
  __shared__ __align__(16) unsigned short Blds[GBN * GLDB];
  int tid = threadIdx.x;
  int lane = tid & 31;
  int wave = tid >> 5;      // 8 waves
  int wm = wave & 3;        // 4 along M (32 rows each)
  int wn = wave >> 2;       // 2 along N (32 cols each)
  int bm0 = blockIdx.x * GBM;
  int bn0 = blockIdx.y * GBN;

  v8f acc[2][2];
#pragma unroll
  for (int i = 0; i < 2; ++i)
#pragma unroll
    for (int j = 0; j < 2; ++j) acc[i][j] = vzero8();

  for (int k0 = 0; k0 < K; k0 += 32) {
    __syncthreads();
#pragma unroll
    for (int it = 0; it < 2; ++it) {           // A tile: 128x32 (straight copy)
      int id = tid + it * 256;
      int row = id >> 2, cc = (id & 3) * 8;
#ifdef HAVE_ASYNC_LDS
      async_cp16(A + (size_t)(bm0 + row) * K + k0 + cc, &Alds[row * GLDA + cc]);
#else
      uint4 v = *(const uint4*)(A + (size_t)(bm0 + row) * K + k0 + cc);
      *(uint4*)(&Alds[row * GLDA + cc]) = v;
#endif
    }
    {                                          // W tile: 32x64, store transposed [n][k]
      int kr = tid >> 3, nc = (tid & 7) * 8;
      uint4 v = *(const uint4*)(W + (size_t)(k0 + kr) * N + bn0 + nc);
      const unsigned short* s = (const unsigned short*)&v;
#pragma unroll
      for (int e = 0; e < 8; ++e) Blds[(nc + e) * GLDB + kr] = s[e];
    }
#ifdef HAVE_ASYNC_LDS
    async_wait0();
#endif
    __syncthreads();
    v16h a0 = load_afrag(Alds, wm * 32, GLDA, 0, lane);
    v16h a1 = load_afrag(Alds, wm * 32 + 16, GLDA, 0, lane);
    v16h b0 = load_bfrag(Blds, wn * 32, GLDB, 0, lane);
    v16h b1 = load_bfrag(Blds, wn * 32 + 16, GLDB, 0, lane);
    acc[0][0] = wmma16(a0, b0, acc[0][0]);
    acc[0][1] = wmma16(a0, b1, acc[0][1]);
    acc[1][0] = wmma16(a1, b0, acc[1][0]);
    acc[1][1] = wmma16(a1, b1, acc[1][1]);
  }

  int hi8 = (lane >> 4) << 3;
  int colc = lane & 15;
#pragma unroll
  for (int tm = 0; tm < 2; ++tm) {
#pragma unroll
    for (int tn = 0; tn < 2; ++tn) {
      int gn = bn0 + wn * 32 + tn * 16 + colc;
      float bv = bias[gn];
#pragma unroll
      for (int r = 0; r < 8; ++r) {
        int gm = bm0 + wm * 32 + tm * 16 + hi8 + r;
        float v = acc[tm][tn][r] + bv;
        if constexpr (MODE == 0) {
          int b = gm >> 12, n = gm & 4095;
          int hh = gn >> 6, d = gn & 63;
          ((unsigned short*)outp)[(((size_t)b * NHEAD + hh) * NTOK + n) * HD + d] =
              f2h(v * scale);
        } else if constexpr (MODE == 2) {
          ((float*)outp)[(size_t)gm * DIMC + gn] = v;
        } else if constexpr (MODE == 3) {
          float gl = 0.5f * v * (1.0f + erff(v * 0.70710678118f));
          ((unsigned short*)outp)[(size_t)gm * N + gn] = f2h(gl);
        } else {
          int b = gm >> 12, n = gm & 4095;
          ((float*)outp)[((size_t)b * DIMC + gn) * NTOK + n] =
              v + aux[(size_t)gm * DIMC + gn];
        }
      }
    }
  }
}

// ---------------- flash attention: per (qblock, batch*head) ----------------
#define ALDK 72
#define PLD  72   // 144B rows -> 16B-aligned A-fragment reloads (ds_load_b128)

__global__ __launch_bounds__(128) void attn_kernel(const unsigned short* __restrict__ Q,
                                                   const unsigned short* __restrict__ Km,
                                                   const unsigned short* __restrict__ Vm,
                                                   unsigned short* __restrict__ O) {
  __shared__ __align__(16) unsigned short Klds[64 * ALDK];
  __shared__ __align__(16) unsigned short Vt[64 * ALDK];
  __shared__ __align__(16) unsigned short Plds[4 * 16 * PLD];
  int tid = threadIdx.x, lane = tid & 31, wave = tid >> 5;
  int qb = blockIdx.x;                 // 64 query blocks of 64 rows
  int bh = blockIdx.y;                 // B*NHEAD = 16
  int b = bh >> 3, h = bh & 7;
  const size_t hoff = (size_t)bh * NTOK * HD;
  int q0 = qb * 64 + wave * 16;

  // Q A-fragments (K = 64 -> 2 frags), loaded once, scattered global u32 reads
  v16h qa[2];
  {
    int m = lane & 15, kh = lane >> 4;
    const unsigned int* qp = (const unsigned int*)(Q + hoff + (size_t)(q0 + m) * HD);
    UF f0, f1;
#pragma unroll
    for (int j = 0; j < 8; ++j) {
      int kk = ((j & 4) << 2) + (kh << 3) + ((j & 3) << 1);
      f0.u[j] = qp[kk >> 1];
      f1.u[j] = qp[(kk + 32) >> 1];
    }
    qa[0] = f0.v; qa[1] = f1.v;
  }

  v8f o[4];
#pragma unroll
  for (int i = 0; i < 4; ++i) o[i] = vzero8();
  float runm[8], runl[8];
#pragma unroll
  for (int r = 0; r < 8; ++r) { runm[r] = -1e30f; runl[r] = 0.0f; }

  int hi8 = (lane >> 4) << 3;
  int colc = lane & 15;

  for (int kt = 0; kt < NTOK / 64; ++kt) {
    __syncthreads();
    int k0 = kt * 64;
#pragma unroll
    for (int it = 0; it < 4; ++it) {   // load K (copy) and V (transposed) 64x64 tiles
      int id = tid + it * 128;
      int row = id >> 3, cc = (id & 7) * 8;
#ifdef HAVE_ASYNC_LDS
      async_cp16(Km + hoff + (size_t)(k0 + row) * HD + cc, &Klds[row * ALDK + cc]);
#else
      uint4 kv = *(const uint4*)(Km + hoff + (size_t)(k0 + row) * HD + cc);
      *(uint4*)(&Klds[row * ALDK + cc]) = kv;
#endif
      uint4 vv = *(const uint4*)(Vm + hoff + (size_t)(k0 + row) * HD + cc);
      const unsigned short* s = (const unsigned short*)&vv;
#pragma unroll
      for (int e = 0; e < 8; ++e) Vt[(cc + e) * ALDK + row] = s[e];
    }
#ifdef HAVE_ASYNC_LDS
    async_wait0();
#endif
    __syncthreads();

    // S = (Q*scale) K^T : 16 x 64 per wave
    v8f s[4];
#pragma unroll
    for (int tn = 0; tn < 4; ++tn) {
      s[tn] = vzero8();
#pragma unroll
      for (int ks = 0; ks < 2; ++ks) {
        v16h bf = load_bfrag(Klds, tn * 16, ALDK, ks * 32, lane);
        s[tn] = wmma16(qa[ks], bf, s[tn]);
      }
    }

    // online softmax (rows live in 16-lane groups; reduce with xor ds_swizzle 1..8)
#pragma unroll
    for (int r = 0; r < 8; ++r) {
      float mx = fmaxf(fmaxf(s[0][r], s[1][r]), fmaxf(s[2][r], s[3][r]));
      mx = fmaxf(mx, SWZ_XOR_F(mx, 1));
      mx = fmaxf(mx, SWZ_XOR_F(mx, 2));
      mx = fmaxf(mx, SWZ_XOR_F(mx, 4));
      mx = fmaxf(mx, SWZ_XOR_F(mx, 8));
      float nm = fmaxf(runm[r], mx);
      float al = __expf(runm[r] - nm);
      runm[r] = nm;
      float rs = 0.0f;
#pragma unroll
      for (int tn = 0; tn < 4; ++tn) {
        float p = __expf(s[tn][r] - nm);
        s[tn][r] = p;
        rs += p;
      }
      rs += SWZ_XOR_F(rs, 1);
      rs += SWZ_XOR_F(rs, 2);
      rs += SWZ_XOR_F(rs, 4);
      rs += SWZ_XOR_F(rs, 8);
      runl[r] = runl[r] * al + rs;
#pragma unroll
      for (int ot = 0; ot < 4; ++ot) o[ot][r] *= al;
    }

    // re-layout P (C-frag -> A-frag) via per-wave LDS
    unsigned short* pw = Plds + wave * 16 * PLD;
#pragma unroll
    for (int tn = 0; tn < 4; ++tn)
#pragma unroll
      for (int r = 0; r < 8; ++r)
        pw[(hi8 + r) * PLD + tn * 16 + colc] = f2h(s[tn][r]);

    // O += P V
#pragma unroll
    for (int ks = 0; ks < 2; ++ks) {
      v16h pa = load_afrag(pw, 0, PLD, ks * 32, lane);
#pragma unroll
      for (int ot = 0; ot < 4; ++ot) {
        v16h vb = load_bfrag(Vt, ot * 16, ALDK, ks * 32, lane);
        o[ot] = wmma16(pa, vb, o[ot]);
      }
    }
  }

  // normalize and store attention output as f16 [B,N,C]
#pragma unroll
  for (int ot = 0; ot < 4; ++ot)
#pragma unroll
    for (int r = 0; r < 8; ++r) {
      float val = o[ot][r] / runl[r];
      int row = q0 + hi8 + r;
      O[((size_t)b * NTOK + row) * DIMC + h * HD + ot * 16 + colc] = f2h(val);
    }
}

// ---------------- LayerNorm fused with first residual ----------------
__global__ __launch_bounds__(256) void ln_kernel(const float* __restrict__ y,
                                                 const float* __restrict__ x,
                                                 const float* __restrict__ gamma,
                                                 const float* __restrict__ beta,
                                                 unsigned short* __restrict__ ln) {
  int wave = threadIdx.x >> 5, lane = threadIdx.x & 31;
  int row = blockIdx.x * 8 + wave;     // [0, 8192)
  int b = row >> 12, n = row & 4095;
  float hv[16];
  float sum = 0.0f, sq = 0.0f;
#pragma unroll
  for (int i = 0; i < 16; ++i) {
    int c = i * 32 + lane;
    float v = y[(size_t)row * DIMC + c] + x[((size_t)b * DIMC + c) * NTOK + n];
    hv[i] = v;
    sum += v;
    sq += v * v;
  }
  sum += SWZ_XOR_F(sum, 1);  sq += SWZ_XOR_F(sq, 1);
  sum += SWZ_XOR_F(sum, 2);  sq += SWZ_XOR_F(sq, 2);
  sum += SWZ_XOR_F(sum, 4);  sq += SWZ_XOR_F(sq, 4);
  sum += SWZ_XOR_F(sum, 8);  sq += SWZ_XOR_F(sq, 8);
  sum += SWZ_XOR_F(sum, 16); sq += SWZ_XOR_F(sq, 16);
  float mu = sum * (1.0f / 512.0f);
  float var = sq * (1.0f / 512.0f) - mu * mu;
  float rstd = rsqrtf(var + 1e-5f);
#pragma unroll
  for (int i = 0; i < 16; ++i) {
    int c = i * 32 + lane;
    ln[(size_t)row * DIMC + c] = f2h((hv[i] - mu) * rstd * gamma[c] + beta[c]);
  }
}

// ---------------- host orchestration ----------------
extern "C" void kernel_launch(void* const* d_in, const int* in_sizes, int n_in,
                              void* d_out, int out_size, void* d_ws, size_t ws_size,
                              hipStream_t stream) {
  const float* x  = (const float*)d_in[0];
  const float* wq = (const float*)d_in[1];
  const float* bq = (const float*)d_in[2];
  const float* wk = (const float*)d_in[3];
  const float* bk = (const float*)d_in[4];
  const float* wv = (const float*)d_in[5];
  const float* bv = (const float*)d_in[6];
  const float* wo = (const float*)d_in[7];
  const float* bo = (const float*)d_in[8];
  const float* w1 = (const float*)d_in[9];
  const float* b1 = (const float*)d_in[10];
  const float* w2 = (const float*)d_in[11];
  const float* b2 = (const float*)d_in[12];
  const float* gamma = (const float*)d_in[13];
  const float* beta  = (const float*)d_in[14];
  float* out = (float*)d_out;

  char* ws = (char*)d_ws;
  size_t off = 0;
  auto alloc = [&](size_t bytes) -> char* {
    char* p = ws + off;
    off += (bytes + 255) & ~(size_t)255;
    return p;
  };
  unsigned short* xf  = (unsigned short*)alloc((size_t)MROWS * DIMC * 2);
  unsigned short* wqf = (unsigned short*)alloc((size_t)DIMC * DIMC * 2);
  unsigned short* wkf = (unsigned short*)alloc((size_t)DIMC * DIMC * 2);
  unsigned short* wvf = (unsigned short*)alloc((size_t)DIMC * DIMC * 2);
  unsigned short* wof = (unsigned short*)alloc((size_t)DIMC * DIMC * 2);
  unsigned short* w1f = (unsigned short*)alloc((size_t)DIMC * 1024 * 2);
  unsigned short* w2f = (unsigned short*)alloc((size_t)1024 * DIMC * 2);
  unsigned short* qh  = (unsigned short*)alloc((size_t)MROWS * DIMC * 2);
  unsigned short* khb = (unsigned short*)alloc((size_t)MROWS * DIMC * 2);
  unsigned short* vhb = (unsigned short*)alloc((size_t)MROWS * DIMC * 2);
  unsigned short* ao  = (unsigned short*)alloc((size_t)MROWS * DIMC * 2);
  float*          yb  = (float*)alloc((size_t)MROWS * DIMC * 4);
  unsigned short* lnb = (unsigned short*)alloc((size_t)MROWS * DIMC * 2);
  unsigned short* gb  = (unsigned short*)alloc((size_t)MROWS * 1024 * 2);

  int nw = DIMC * DIMC;
  cvt_f32_f16<<<(nw + 255) / 256, 256, 0, stream>>>(wq, wqf, nw);
  cvt_f32_f16<<<(nw + 255) / 256, 256, 0, stream>>>(wk, wkf, nw);
  cvt_f32_f16<<<(nw + 255) / 256, 256, 0, stream>>>(wv, wvf, nw);
  cvt_f32_f16<<<(nw + 255) / 256, 256, 0, stream>>>(wo, wof, nw);
  int nm = DIMC * 1024;
  cvt_f32_f16<<<(nm + 255) / 256, 256, 0, stream>>>(w1, w1f, nm);
  cvt_f32_f16<<<(nm + 255) / 256, 256, 0, stream>>>(w2, w2f, nm);

  xpose_f16<<<dim3(NTOK / 32, DIMC / 32, BATCH), dim3(32, 32), 0, stream>>>(x, xf);

  dim3 blk(256);
  dim3 g512(MROWS / GBM, DIMC / GBN);
  gemm16<0><<<g512, blk, 0, stream>>>(xf, wqf, bq, qh, nullptr, MROWS, DIMC, DIMC, 0.125f);
  gemm16<0><<<g512, blk, 0, stream>>>(xf, wkf, bk, khb, nullptr, MROWS, DIMC, DIMC, 1.0f);
  gemm16<0><<<g512, blk, 0, stream>>>(xf, wvf, bv, vhb, nullptr, MROWS, DIMC, DIMC, 1.0f);

  attn_kernel<<<dim3(NTOK / 64, BATCH * NHEAD), 128, 0, stream>>>(qh, khb, vhb, ao);

  gemm16<2><<<g512, blk, 0, stream>>>(ao, wof, bo, yb, nullptr, MROWS, DIMC, DIMC, 1.0f);

  ln_kernel<<<MROWS / 8, 256, 0, stream>>>(yb, x, gamma, beta, lnb);

  gemm16<3><<<dim3(MROWS / GBM, 1024 / GBN), blk, 0, stream>>>(lnb, w1f, b1, gb, nullptr,
                                                               MROWS, 1024, DIMC, 1.0f);
  gemm16<4><<<g512, blk, 0, stream>>>(gb, w2f, b2, out, yb, MROWS, DIMC, 1024, 1.0f);
}
